// TLSGCNCNN_49263274885786
// MI455X (gfx1250) — matrix operand order repaired and checked by reading, not compiled, not run on hardware
//
#include <hip/hip_runtime.h>
#include <hip/hip_bf16.h>

typedef __bf16 v16bf __attribute__((ext_vector_type(16)));
typedef float  v8f   __attribute__((ext_vector_type(8)));

#define NNODE 2048
#define EMB   16

// ---------------------------------------------------------------------------
// sup = softmax(relu(E E^T)) row-wise, emitted directly as bf16 for WMMA use.
// ---------------------------------------------------------------------------
__global__ __launch_bounds__(256) void gram_softmax_kernel(
    const float* __restrict__ E, __hip_bfloat16* __restrict__ supB)
{
    __shared__ float row[NNODE];
    __shared__ float red[256];
    __shared__ float en[EMB];
    const int n = blockIdx.x, t = threadIdx.x;
    if (t < EMB) en[t] = E[n * EMB + t];
    __syncthreads();

    float lmax = -1e30f;
    for (int m = t; m < NNODE; m += 256) {
        const float* em = E + m * EMB;
        float d = 0.f;
#pragma unroll
        for (int j = 0; j < EMB; ++j) d += en[j] * em[j];
        d = fmaxf(d, 0.f);
        row[m] = d;
        lmax = fmaxf(lmax, d);
    }
    red[t] = lmax; __syncthreads();
    for (int s = 128; s > 0; s >>= 1) {
        if (t < s) red[t] = fmaxf(red[t], red[t + s]);
        __syncthreads();
    }
    const float mx = red[0];
    __syncthreads();
    float lsum = 0.f;
    for (int m = t; m < NNODE; m += 256) {
        float e = __expf(row[m] - mx);
        row[m] = e;
        lsum += e;
    }
    red[t] = lsum; __syncthreads();
    for (int s = 128; s > 0; s >>= 1) {
        if (t < s) red[t] += red[t + s];
        __syncthreads();
    }
    const float inv = 1.f / red[0];
    for (int m = t; m < NNODE; m += 256)
        supB[(size_t)n * NNODE + m] = __float2bfloat16(row[m] * inv);
}

// ---------------------------------------------------------------------------
// Pack x (B,N,64) f32 -> Xp (N, B*64) bf16  (GEMM B operand, row = contraction m)
// ---------------------------------------------------------------------------
__global__ __launch_bounds__(256) void pack_x_kernel(
    const float* __restrict__ x, __hip_bfloat16* __restrict__ xp)
{
    const int idx = blockIdx.x * 256 + threadIdx.x; // < 32*2048*64
    const int c = idx & 63;
    const int m = (idx >> 6) & (NNODE - 1);
    const int b = idx >> 17;
    xp[(size_t)m * 2048 + b * 64 + c] = __float2bfloat16(x[idx]);
}

// ---------------------------------------------------------------------------
// xwt[b,m,i] = sum_t T[t]*x_window[b,t,m,i]; write f32 (k=0 term of final
// contraction) and bf16 packed (N, B*32) for the GEMMs.  12x flop reduction
// versus contracting T last, as the reference does.
// ---------------------------------------------------------------------------
__global__ __launch_bounds__(256) void pack_xw_kernel(
    const float* __restrict__ xw, const float* __restrict__ T,
    float* __restrict__ xwt0, __hip_bfloat16* __restrict__ xwp)
{
    const int idx = blockIdx.x * 256 + threadIdx.x; // < 32*2048*32
    const int i = idx & 31;
    const int m = (idx >> 5) & (NNODE - 1);
    const int b = idx >> 16;
    float s = 0.f;
#pragma unroll
    for (int tt = 0; tt < 12; ++tt)
        s += T[tt] * xw[(((size_t)(b * 12 + tt)) * NNODE + m) * 32 + i];
    const size_t o = (size_t)m * 1024 + b * 32 + i;
    xwt0[o] = s;
    xwp[o]  = __float2bfloat16(s);
}

__global__ __launch_bounds__(256) void convert_f32_bf16_kernel(
    const float* __restrict__ in, __hip_bfloat16* __restrict__ out, int n)
{
    const int idx = blockIdx.x * 256 + threadIdx.x;
    if (idx < n) out[idx] = __float2bfloat16(in[idx]);
}

// ---------------------------------------------------------------------------
// bf16 WMMA GEMM: C(MxN,f32) = A(MxK,bf16,row-major) * B(KxN,bf16,row-major)
// 128x128 block tile, 8 waves, each wave 32(M)x64(N) via 2x4 wmma frags.
// Double-buffered LDS: iteration i reads buffer i&1 while staging buffer
// (i+1)&1, so global loads + LDS stores co-execute with the 8 XDL WMMAs.
// LDS row stride 40 bf16 (80B): 16B-aligned rows, conflict-free b128 frag
// reads.  B is transposed on LDS store so B fragments (K 8g..8g+7 and 8g+16..8g+23 per
// column, per the CDNA5 B-matrix layout) are two contiguous 16B ds reads.
// ---------------------------------------------------------------------------
#define LDS_STR 40
__global__ __launch_bounds__(256) void gemm_bf16_kernel(
    const __hip_bfloat16* __restrict__ A,
    const __hip_bfloat16* __restrict__ B,
    float* __restrict__ C, int M, int N, int K)
{
    __shared__ __hip_bfloat16 As [2][128][LDS_STR];
    __shared__ __hip_bfloat16 Bts[2][128][LDS_STR];

    const int t    = threadIdx.x;
    const int bm   = blockIdx.y * 128;
    const int bn   = blockIdx.x * 128;
    const int wave = t >> 5, lane = t & 31;
    const int wm   = (wave & 3) * 32;   // 4 waves along M
    const int wn   = (wave >> 2) * 64;  // 2 waves along N
    const int g    = lane >> 4;         // lane group (K half selector)
    const int ln   = lane & 15;         // row (A) / col (B/D) index

    v8f acc[2][4];
#pragma unroll
    for (int a = 0; a < 2; ++a)
#pragma unroll
        for (int b2 = 0; b2 < 4; ++b2) acc[a][b2] = (v8f)(0.f);

    // staging coordinates
    const int arow = t >> 1, acol = (t & 1) * 16;   // A: 128 rows x 32 K
    const int bk   = t >> 3, bn0  = (t & 7) * 16;   // B: 32 K rows x 128 cols
    const __hip_bfloat16* aSrc = A + (size_t)(bm + arow) * K + acol;
    const __hip_bfloat16* bSrc = B + (size_t)bk * N + bn + bn0;

    // prologue: stage K-tile 0 into buffer 0
    {
        *(float4*)(&As[0][arow][acol])     = *(const float4*)(aSrc);
        *(float4*)(&As[0][arow][acol + 8]) = *(const float4*)(aSrc + 8);
        __hip_bfloat16 tmp[16];
        *(float4*)(tmp)     = *(const float4*)(bSrc);
        *(float4*)(tmp + 8) = *(const float4*)(bSrc + 8);
#pragma unroll
        for (int j = 0; j < 16; ++j) Bts[0][bn0 + j][bk] = tmp[j];
    }
    __syncthreads();

    for (int k0 = 0; k0 < K; k0 += 32) {
        const int cur = (k0 >> 5) & 1;
        const int nxt = cur ^ 1;

        if (k0 + 32 < K) {  // stage next K-tile into the other buffer
            const __hip_bfloat16* sa = aSrc + (k0 + 32);
            const __hip_bfloat16* sb = bSrc + (size_t)(k0 + 32) * N;
            *(float4*)(&As[nxt][arow][acol])     = *(const float4*)(sa);
            *(float4*)(&As[nxt][arow][acol + 8]) = *(const float4*)(sa + 8);
            __hip_bfloat16 tmp[16];
            *(float4*)(tmp)     = *(const float4*)(sb);
            *(float4*)(tmp + 8) = *(const float4*)(sb + 8);
#pragma unroll
            for (int j = 0; j < 16; ++j) Bts[nxt][bn0 + j][bk] = tmp[j];
        }

        v16bf afrag[2], bfrag[4];
#pragma unroll
        for (int tm = 0; tm < 2; ++tm) {
            const int m = wm + tm * 16 + ln;
            float4* pv = reinterpret_cast<float4*>(&afrag[tm]);
            pv[0] = *(const float4*)(&As[cur][m][8 * g]);       // K = 8g..8g+7
            pv[1] = *(const float4*)(&As[cur][m][8 * g + 16]);  // K = 8g+16..8g+23
        }
#pragma unroll
        for (int tn = 0; tn < 4; ++tn) {
            const int n = wn + tn * 16 + ln;
            float4* pv = reinterpret_cast<float4*>(&bfrag[tn]);
            pv[0] = *(const float4*)(&Bts[cur][n][8 * g]);      // K = 8g..8g+7
            pv[1] = *(const float4*)(&Bts[cur][n][8 * g + 16]); // K = 8g+16..8g+23
        }
#pragma unroll
        for (int tm = 0; tm < 2; ++tm)
#pragma unroll
            for (int tn = 0; tn < 4; ++tn)
                acc[tm][tn] = __builtin_amdgcn_wmma_f32_16x16x32_bf16(
                    false, afrag[tm], false, bfrag[tn],
                    (short)0, acc[tm][tn], false, false);

        // one barrier per iteration: buffer written this iter is first read
        // after it; buffer read this iter is first overwritten after the next.
        __syncthreads();
    }

    // D layout: VGPR r -> row (r + 8g), lane&15 -> col
    float* cp = C + (size_t)(bm + wm + 8 * g) * N + (bn + wn + ln);
#pragma unroll
    for (int tm = 0; tm < 2; ++tm)
#pragma unroll
        for (int r = 0; r < 8; ++r) {
            const size_t roff = (size_t)(tm * 16 + r) * N;
#pragma unroll
            for (int tn = 0; tn < 4; ++tn)
                cp[roff + tn * 16] = acc[tm][tn][r];
        }
}

// ---------------------------------------------------------------------------
// CNN gate (tiny: ~45M MACs total)
// ---------------------------------------------------------------------------
__global__ __launch_bounds__(256) void conv1_kernel(
    const float* __restrict__ PI, const float* __restrict__ w,
    const float* __restrict__ bsc, float* __restrict__ out)
{
    const int idx = blockIdx.x * 256 + threadIdx.x; // 32*8*49*49
    const int ox = idx % 49, oy = (idx / 49) % 49;
    const int oc = (idx / (49 * 49)) % 8, b = idx / (8 * 49 * 49);
    const float* src = PI + (size_t)b * 10000;
    const float* wk  = w + oc * 9;
    float s = 0.f;
#pragma unroll
    for (int ky = 0; ky < 3; ++ky)
#pragma unroll
        for (int kx = 0; kx < 3; ++kx)
            s += wk[ky * 3 + kx] * src[(oy * 2 + ky) * 100 + ox * 2 + kx];
    out[idx] = fmaxf(s + bsc[oc], 0.f);
}

__global__ __launch_bounds__(256) void conv2_kernel(
    const float* __restrict__ in, const float* __restrict__ w,
    const float* __restrict__ bsc, float* __restrict__ out)
{
    const int idx = blockIdx.x * 256 + threadIdx.x; // 32*16*24*24
    const int ox = idx % 24, oy = (idx / 24) % 24;
    const int oc = (idx / (24 * 24)) % 16, b = idx / (16 * 24 * 24);
    float s = 0.f;
#pragma unroll
    for (int ic = 0; ic < 8; ++ic) {
        const float* src = in + ((size_t)(b * 8 + ic)) * 49 * 49;
        const float* wk  = w + ((oc * 8 + ic) * 9);
#pragma unroll
        for (int ky = 0; ky < 3; ++ky)
#pragma unroll
            for (int kx = 0; kx < 3; ++kx)
                s += wk[ky * 3 + kx] * src[(oy * 2 + ky) * 49 + ox * 2 + kx];
    }
    out[idx] = fmaxf(s + bsc[oc], 0.f);
}

__global__ __launch_bounds__(64) void fc_kernel(
    const float* __restrict__ h, const float* __restrict__ w,
    const float* __restrict__ bias, float* __restrict__ topo)
{
    const int b = blockIdx.x >> 6, o = blockIdx.x & 63, t = threadIdx.x;
    const float* hb = h + (size_t)b * 9216;
    const float* wo = w + (size_t)o * 9216;
    float s = 0.f;
    for (int j = t; j < 9216; j += 64) s += hb[j] * wo[j];
    __shared__ float red[64];
    red[t] = s; __syncthreads();
    for (int k = 32; k > 0; k >>= 1) {
        if (t < k) red[t] += red[t + k];
        __syncthreads();
    }
    if (t == 0) topo[blockIdx.x] = red[0] + bias[o];
}

// ---------------------------------------------------------------------------
// Final per-node contractions.  W_n regenerated in LDS (avoids a 100MB
// weights buffer); xg values read via all-lanes-same-address broadcast loads
// (a wave spans 32 'o' values of one 'b').  Writes gated + biased output.
// ---------------------------------------------------------------------------
__global__ __launch_bounds__(256) void final_g_kernel(
    const float* __restrict__ E, const float* __restrict__ Wp,      // (16,12288)
    const float* __restrict__ bias_pool,                            // (16,128)
    const float* __restrict__ x,                                    // (32,2048,64)
    const float* __restrict__ xg1, const float* __restrict__ xg2,   // (2048,2048)
    const float* __restrict__ topo,                                 // (32,64)
    float* __restrict__ out)                                        // (32,2048,128)
{
    __shared__ float Wg[3 * 64 * 64]; // 48KB
    __shared__ float en[EMB];
    __shared__ float bsh[64];
    const int n = blockIdx.x, t = threadIdx.x;
    if (t < EMB) en[t] = E[n * EMB + t];
    __syncthreads();
    for (int idx = t; idx < 12288; idx += 256) {
        float s = 0.f;
#pragma unroll
        for (int d = 0; d < EMB; ++d) s += en[d] * Wp[d * 12288 + idx];
        Wg[idx] = s;
    }
    if (t < 64) {
        float s = 0.f;
#pragma unroll
        for (int d = 0; d < EMB; ++d) s += en[d] * bias_pool[d * 128 + t];
        bsh[t] = s;
    }
    __syncthreads();
#pragma unroll
    for (int r = 0; r < 8; ++r) {
        const int idx = t + r * 256;            // 0..2047
        const int b = idx >> 6, o = idx & 63;
        const float* x0 = x   + ((size_t)b * NNODE + n) * 64;
        const float* x1 = xg1 + (size_t)n * 2048 + b * 64;
        const float* x2 = xg2 + (size_t)n * 2048 + b * 64;
        float s = 0.f;
        for (int i = 0; i < 64; ++i) {
            s += x0[i] * Wg[i * 64 + o];
            s += x1[i] * Wg[4096 + i * 64 + o];
            s += x2[i] * Wg[8192 + i * 64 + o];
        }
        out[((size_t)b * NNODE + n) * 128 + o] = s * topo[b * 64 + o] + bsh[o];
    }
}

__global__ __launch_bounds__(256) void final_w_kernel(
    const float* __restrict__ E, const float* __restrict__ Wpw,     // (16,6144)
    const float* __restrict__ bias_pool,
    const float* __restrict__ xw0, const float* __restrict__ xw1,
    const float* __restrict__ xw2,                                  // (2048,1024)
    const float* __restrict__ topo,
    float* __restrict__ out)
{
    __shared__ float Ww[3 * 32 * 64]; // 24KB
    __shared__ float en[EMB];
    __shared__ float bsh[64];
    const int n = blockIdx.x, t = threadIdx.x;
    if (t < EMB) en[t] = E[n * EMB + t];
    __syncthreads();
    for (int idx = t; idx < 6144; idx += 256) {
        float s = 0.f;
#pragma unroll
        for (int d = 0; d < EMB; ++d) s += en[d] * Wpw[d * 6144 + idx];
        Ww[idx] = s;
    }
    if (t < 64) {
        float s = 0.f;
#pragma unroll
        for (int d = 0; d < EMB; ++d) s += en[d] * bias_pool[d * 128 + 64 + t];
        bsh[t] = s;
    }
    __syncthreads();
#pragma unroll
    for (int r = 0; r < 8; ++r) {
        const int idx = t + r * 256;
        const int b = idx >> 6, o = idx & 63;
        const size_t base = (size_t)n * 1024 + b * 32;
        const float* p0 = xw0 + base;
        const float* p1 = xw1 + base;
        const float* p2 = xw2 + base;
        float s = 0.f;
        for (int i = 0; i < 32; ++i) {
            s += p0[i] * Ww[i * 64 + o];
            s += p1[i] * Ww[2048 + i * 64 + o];
            s += p2[i] * Ww[4096 + i * 64 + o];
        }
        out[((size_t)b * NNODE + n) * 128 + 64 + o] = s * topo[b * 64 + o] + bsh[o];
    }
}

// ---------------------------------------------------------------------------
extern "C" void kernel_launch(void* const* d_in, const int* in_sizes, int n_in,
                              void* d_out, int out_size, void* d_ws, size_t ws_size,
                              hipStream_t stream) {
    const float* x    = (const float*)d_in[0];
    const float* xwin = (const float*)d_in[1];
    const float* E    = (const float*)d_in[2];
    const float* PI   = (const float*)d_in[3];
    const float* Wp   = (const float*)d_in[4];
    const float* Wpw  = (const float*)d_in[5];
    const float* bp   = (const float*)d_in[6];
    const float* T    = (const float*)d_in[7];
    const float* c1w  = (const float*)d_in[8];
    const float* c1b  = (const float*)d_in[9];
    const float* c2w  = (const float*)d_in[10];
    const float* c2b  = (const float*)d_in[11];
    const float* fcw  = (const float*)d_in[12];
    const float* fcb  = (const float*)d_in[13];
    float* out = (float*)d_out;

    char* ws = (char*)d_ws;
    __hip_bfloat16* supB  = (__hip_bfloat16*)(ws + 0);          //  8 MB
    __hip_bfloat16* xpack = (__hip_bfloat16*)(ws + 8388608);    //  8 MB
    __hip_bfloat16* xwp   = (__hip_bfloat16*)(ws + 16777216);   //  4 MB
    float*          xwt0  = (float*)        (ws + 20971520);    //  8 MB
    float*          xg1   = (float*)        (ws + 29360128);    // 16 MB
    __hip_bfloat16* xg1b  = (__hip_bfloat16*)(ws + 46137344);   //  8 MB
    float*          xg2   = (float*)        (ws + 54525952);    // 16 MB
    float*          xw1   = (float*)        (ws + 71303168);    //  8 MB
    __hip_bfloat16* xw1b  = (__hip_bfloat16*)(ws + 79691776);   //  4 MB
    float*          xw2   = (float*)        (ws + 83886080);    //  8 MB
    float*          c1o   = (float*)        (ws + 92274688);    //  2.5 MB
    float*          c2o   = (float*)        (ws + 94733312);    //  1.2 MB
    float*          topo  = (float*)        (ws + 95912960);    //  8 KB

    // sup = softmax(relu(E E^T)), bf16
    gram_softmax_kernel<<<NNODE, 256, 0, stream>>>(E, supB);
    // pack GEMM B operands
    pack_x_kernel <<<16384, 256, 0, stream>>>(x, xpack);
    pack_xw_kernel<<<8192, 256, 0, stream>>>(xwin, T, xwt0, xwp);
    // CNN gate
    conv1_kernel<<<2401, 256, 0, stream>>>(PI, c1w, c1b, c1o);
    conv2_kernel<<<1152, 256, 0, stream>>>(c1o, c2w, c2b, c2o);
    fc_kernel   <<<2048, 64, 0, stream>>>(c2o, fcw, fcb, topo);

    // supports GEMMs (sup^2 never materialized: sup^2 @ X == sup @ (sup @ X))
    gemm_bf16_kernel<<<dim3(16, 16), 256, 0, stream>>>(supB, xpack, xg1, 2048, 2048, 2048);
    convert_f32_bf16_kernel<<<16384, 256, 0, stream>>>(xg1, xg1b, 2048 * 2048);
    gemm_bf16_kernel<<<dim3(16, 16), 256, 0, stream>>>(supB, xg1b, xg2, 2048, 2048, 2048);
    gemm_bf16_kernel<<<dim3(8, 16), 256, 0, stream>>>(supB, xwp, xw1, 2048, 1024, 2048);
    convert_f32_bf16_kernel<<<8192, 256, 0, stream>>>(xw1, xw1b, 2048 * 1024);
    gemm_bf16_kernel<<<dim3(8, 16), 256, 0, stream>>>(supB, xw1b, xw2, 2048, 1024, 2048);

    // per-node contractions + gate + bias -> output
    final_g_kernel<<<NNODE, 256, 0, stream>>>(E, Wp, bp, x, xg1, xg2, topo, out);
    final_w_kernel<<<NNODE, 256, 0, stream>>>(E, Wpw, bp, xwt0, xw1, xw2, topo, out);
}
